// LSTMClassifier_68788196212805
// MI455X (gfx1250) — matrix operand order repaired and checked by reading, not compiled, not run on hardware
//
#include <hip/hip_runtime.h>

// ---------------------------------------------------------------------------
// Fused LSTM (B=4096,T=256,F=32,H=128) + LayerNorm + linear head for gfx1250.
//
// Design (MI455X reasoning):
//  * 172 GFLOP of gate GEMMs vs 134 MB of x traffic: compute/latency bound
//    sequential scan -> weights resident in LDS (f16 W_ih||W_hh = 160KB of
//    the 320KB WGP LDS), cell state in VGPRs, whole T=256 scan in one WG.
//  * One WG (256 thr = 8 wave32) per 32 batch rows; 4 waves per 16-row tile,
//    one gate (128 of 512 gate cols) per wave.
//  * Per step/wave: 5 K-chunks x 8 N-tiles of v_wmma_f32_16x16x32_f16
//    (A = [x_t || h_{t-1}] f16 from LDS, B = resident f16 weights), with
//    B-fragments software-pipelined one N-tile ahead so LDS latency is
//    overlapped with WMMA issue.
//  * x_{t+1} is prefetched by the Tensor Data Mover (tensor_load_to_lds,
//    TENSORcnt) into a double-buffered LDS stage while step t computes,
//    removing HBM latency from the serial timestep chain.
// ---------------------------------------------------------------------------

#define DEV_INLINE __device__ __forceinline__

typedef __attribute__((ext_vector_type(16))) _Float16 v16h;
typedef __attribute__((ext_vector_type(8)))  _Float16 v8h;
typedef __attribute__((ext_vector_type(4)))  _Float16 v4h;
typedef __attribute__((ext_vector_type(8)))  float    v8f;
typedef __attribute__((ext_vector_type(4)))  float    v4f;
typedef __attribute__((ext_vector_type(4)))  unsigned int v4u;
typedef __attribute__((ext_vector_type(8)))  int      v8i;
typedef __attribute__((ext_vector_type(4)))  int      v4i;

struct H16 { v8h lo, hi; };
struct F8  { v4f lo, hi; };

static constexpr int Bsz = 4096;
static constexpr int T   = 256;
static constexpr int F   = 32;
static constexpr int H   = 128;
static constexpr int G   = 4 * H;     // 512 gate columns
static constexpr int K   = F + H;     // 160 combined K dim

// ---- LDS layout (bytes) ----------------------------------------------------
static constexpr int OFF_WB    = 0;                        // f16 [512][160]
static constexpr int SZ_WB     = G * K * 2;                // 163840
static constexpr int OFF_BIAS  = OFF_WB + SZ_WB;           // f32 [512]
static constexpr int SZ_BIAS   = G * 4;                    // 2048
static constexpr int OFF_ABUF  = OFF_BIAS + SZ_BIAS;       // f16 [32][160]
static constexpr int SZ_ABUF   = 32 * K * 2;               // 10240
static constexpr int OFF_GBUF  = OFF_ABUF + SZ_ABUF;       // f32 [2][512][16]
static constexpr int SZ_GBUF   = 2 * G * 16 * 4;           // 65536
static constexpr int OFF_HLAST = OFF_GBUF + SZ_GBUF;       // f32 [2][16][128]
static constexpr int SZ_HLAST  = 2 * 16 * H * 4;           // 16384
static constexpr int OFF_XSTG  = OFF_HLAST + SZ_HLAST;     // f32 [2][32][32]
static constexpr int SZ_XSTG   = 2 * 32 * F * 4;           // 8192
static constexpr int SMEM_BYTES = OFF_XSTG + SZ_XSTG;      // 266240 (< 320KB)

DEV_INLINE float fsig(float x) {
    return __builtin_amdgcn_rcpf(1.0f + __expf(-x));
}
DEV_INLINE float ftanh(float x) {
    // tanh(x) = 1 - 2/(exp(2x)+1); saturates correctly at +-1 for |x| large.
    return 1.0f - 2.0f * __builtin_amdgcn_rcpf(1.0f + __expf(2.0f * x));
}

// Issue one TDM 2D tile load: 32 rows x 32 f32, row stride T*F elements,
// from global `gsrc` into LDS byte offset `lds_off`.
DEV_INLINE void tdm_load_x(const float* gsrc, unsigned int lds_off) {
    const unsigned long long ga = (unsigned long long)(uintptr_t)gsrc;
    v4u g0;
    g0.x = 1u;                                   // count=1, user descriptor
    g0.y = lds_off;                              // lds_addr
    g0.z = (unsigned int)ga;                     // global_addr[31:0]
    g0.w = (unsigned int)(ga >> 32) | (2u << 30);// global_addr[56:32] | type=2
    v8i g1;
    g1[0] = (int)(2u << 16);                     // data_size = 4B
    g1[1] = (int)(32u << 16);                    // tensor_dim0 = 32 (bits 79:48)
    g1[2] = (int)(32u << 16);                    // tensor_dim1 = 32 (bits 111:80)
    g1[3] = (int)(32u << 16);                    // tile_dim0 = 32 (bits 127:112)
    g1[4] = 32;                                  // tile_dim1 = 32
    g1[5] = T * F;                               // tensor_dim0_stride = 8192
    g1[6] = 0;
    g1[7] = 0;
    const v4i z4 = {0, 0, 0, 0};                 // groups 2/3 unused (2D tile)
    const v8i z8 = {0, 0, 0, 0, 0, 0, 0, 0};
    __builtin_amdgcn_tensor_load_to_lds(g0, g1, z4, z4, z8, 0);
}

// Load the 5 K-chunk B-fragments for one 16-wide N-tile.
DEV_INLINE void ldB5(const _Float16* br, v16h bf[5]) {
#pragma unroll
    for (int c = 0; c < 5; ++c) {
        H16 hh;
        hh.lo = *(const v8h*)(br + c * 32);      // ds_load_b128
        hh.hi = *(const v8h*)(br + c * 32 + 16); // ds_load_b128
        bf[c] = __builtin_bit_cast(v16h, hh);
    }
}

__global__ __launch_bounds__(256)
void lstm_ln_head_kernel(const float* __restrict__ x,
                         const float* __restrict__ W_ih,
                         const float* __restrict__ W_hh,
                         const float* __restrict__ b_ih,
                         const float* __restrict__ b_hh,
                         const float* __restrict__ ln_g,
                         const float* __restrict__ ln_b,
                         const float* __restrict__ W_head,
                         const float* __restrict__ b_head,
                         float* __restrict__ out)
{
    extern __shared__ char smem[];
    _Float16* wB    = (_Float16*)(smem + OFF_WB);
    float*    bias  = (float*)   (smem + OFF_BIAS);
    _Float16* abuf  = (_Float16*)(smem + OFF_ABUF);
    float*    gbuf  = (float*)   (smem + OFF_GBUF);
    float*    hlast = (float*)   (smem + OFF_HLAST);
    float*    xstg  = (float*)   (smem + OFF_XSTG);
    // LDS byte offset of xstg (generic shared addr low 32 bits == LDS offset)
    const unsigned int xstg_off = (unsigned int)(uintptr_t)(void*)xstg;

    const int tid    = threadIdx.x;
    const int lane   = tid & 31;
    const int wave   = tid >> 5;       // 0..7
    const int tile   = wave >> 2;      // 0/1: which 16-row batch tile
    const int wl     = wave & 3;       // wave-in-tile == gate index
    const int lane16 = lane & 15;
    const bool hiHalf = lane >= 16;
    const int b0     = blockIdx.x * 32;
    const float* xbase = x + (size_t)b0 * (T * F);

    // Prefetch x_0 tile via TDM while we stage weights.
    if (wave == 0) tdm_load_x(xbase, xstg_off);

    // ---- Stage weights into LDS as f16 pane wB[n][k] (k<32 -> W_ih, else W_hh)
    for (int idx = tid; idx < G * K; idx += 256) {
        const int n = idx / K, k = idx - n * K;
        const float v = (k < F) ? W_ih[n * F + k] : W_hh[n * H + (k - F)];
        wB[idx] = (_Float16)v;
    }
    for (int n = tid; n < G; n += 256) bias[n] = b_ih[n] + b_hh[n];
    // zero h_{-1} region of the A-buffer
    for (int idx = tid; idx < 32 * H; idx += 256) {
        const int r  = idx / H;
        const int c2 = idx % H;
        abuf[r * K + F + c2] = (_Float16)0.0f;
    }
    if (wave == 0) __builtin_amdgcn_s_wait_tensorcnt(0);
    __syncthreads();

    // ---- per-lane constants -------------------------------------------------
    // x staging slot: 256 threads cover 32 rows x 8 dword-groups
    const int sr = tid >> 3;                   // row 0..31 (both tiles)
    const int sc = (tid & 7) * 4;              // float offset within F=32
    const float* xsrc = xstg + sr * F + sc;
    _Float16*    xdst = &abuf[sr * K + sc];

    // element-wise phase: this lane owns h column `ncol` (0..127) of its tile
    const int ncol = wl * 32 + lane;
    const float bi = bias[0 * H + ncol];
    const float bf = bias[1 * H + ncol];
    const float bg = bias[2 * H + ncol];
    const float bo = bias[3 * H + ncol];

    // A-fragment base (16-bit A 16x32 layout: lanes<16 K=[0..7],[16..23];
    // lanes>=16 K=[8..15],[24..31])
    const _Float16* arow  = &abuf[(tile * 16 + lane16) * K + (hiHalf ? 8 : 0)];
    // B-fragment base (lanes<16: N=lane,K=[0..7],[16..23]; lanes>=16: N=lane-16,K=[8..15],[24..31])
    const _Float16* brow0 = &wB[(wl * 128 + lane16) * K + (hiHalf ? 8 : 0)];
    // C/D store base in gbuf[tile][n][m]
    float* gdst = &gbuf[((size_t)tile * G + wl * 128 + lane16) * 16 + (hiHalf ? 8 : 0)];
    // gate read bases for elementwise
    const float* gI = &gbuf[((size_t)tile * G + 0 * H + ncol) * 16];
    const float* gF = &gbuf[((size_t)tile * G + 1 * H + ncol) * 16];
    const float* gG = &gbuf[((size_t)tile * G + 2 * H + ncol) * 16];
    const float* gO = &gbuf[((size_t)tile * G + 3 * H + ncol) * 16];
    _Float16* hdst = &abuf[(tile * 16) * K + F + ncol];

    float cst[16];
#pragma unroll
    for (int m = 0; m < 16; ++m) cst[m] = 0.0f;

    // =========================== time scan ==================================
    for (int t = 0; t < T; ++t) {
        // -- stage x_t (f32 from TDM buffer -> f16) into A-buffer cols [0,32)
        const v4f xv = *(const v4f*)(xsrc + (t & 1) * (32 * F));
        v4h xh;
        xh.x = (_Float16)xv.x; xh.y = (_Float16)xv.y;
        xh.z = (_Float16)xv.z; xh.w = (_Float16)xv.w;
        *(v4h*)xdst = xh;                       // ds_store_b64
        __syncthreads();                        // B1: x_t + h_{t-1} visible

        // -- kick TDM prefetch of x_{t+1} into the other staging buffer
        if (wave == 0 && (t + 1) < T)
            tdm_load_x(xbase + (size_t)(t + 1) * F,
                       xstg_off + (unsigned int)(((t + 1) & 1) * (32 * F * 4)));

        // -- load A fragments once (5 K-chunks), reuse across 8 N-tiles
        v16h afrag[5];
#pragma unroll
        for (int c = 0; c < 5; ++c) {
            H16 hh;
            hh.lo = *(const v8h*)(arow + c * 32);        // ds_load_b128
            hh.hi = *(const v8h*)(arow + c * 32 + 16);   // ds_load_b128
            afrag[c] = __builtin_bit_cast(v16h, hh);
        }

        // -- 8 N-tiles x 5 K-chunks of WMMA, B-frags pipelined one tile ahead
        v16h bcur[5], bnxt[5];
        ldB5(brow0, bcur);
#pragma unroll
        for (int j = 0; j < 8; ++j) {
            if (j < 7) ldB5(brow0 + ((j + 1) * 16) * K, bnxt);
            v8f acc = {0.f, 0.f, 0.f, 0.f, 0.f, 0.f, 0.f, 0.f};
#pragma unroll
            for (int c = 0; c < 5; ++c)
                acc = __builtin_amdgcn_wmma_f32_16x16x32_f16(
                    false, afrag[c], false, bcur[c], (short)0, acc, false, false);
            const F8 st = __builtin_bit_cast(F8, acc);
            float* gd = gdst + (j * 16) * 16;
            *(v4f*)gd       = st.lo;                     // ds_store_b128
            *(v4f*)(gd + 4) = st.hi;                     // ds_store_b128
            if (j < 7) {
#pragma unroll
                for (int c = 0; c < 5; ++c) bcur[c] = bnxt[c];
            }
        }
        __syncthreads();                        // B2: gates visible

        // -- element-wise LSTM cell update for this lane's h column
#pragma unroll
        for (int mb = 0; mb < 4; ++mb) {
            const v4f iv = *(const v4f*)(gI + mb * 4);
            const v4f fv = *(const v4f*)(gF + mb * 4);
            const v4f gv = *(const v4f*)(gG + mb * 4);
            const v4f ov = *(const v4f*)(gO + mb * 4);
#pragma unroll
            for (int q = 0; q < 4; ++q) {
                const int m  = mb * 4 + q;
                const float ig = fsig(iv[q] + bi);
                const float fg = fsig(fv[q] + bf);
                const float gg = ftanh(gv[q] + bg);
                const float og = fsig(ov[q] + bo);
                const float c  = fg * cst[m] + ig * gg;
                cst[m] = c;
                const float h = og * ftanh(c);
                hdst[m * K] = (_Float16)h;              // next step's A operand
                if (t == T - 1)
                    hlast[((size_t)tile * 16 + m) * H + ncol] = h;
            }
        }
        // TDM prefetch must be complete before anyone reads the stage buffer
        if (wave == 0) __builtin_amdgcn_s_wait_tensorcnt(0);
        __syncthreads();                        // B3: h_t + x_{t+1} stage ready
    }

    // ====================== LayerNorm + head =================================
    const v4f lg = *(const v4f*)(ln_g   + lane * 4);
    const v4f lb = *(const v4f*)(ln_b   + lane * 4);
    const v4f wh = *(const v4f*)(W_head + lane * 4);
    const float bh = b_head[0];

#pragma unroll
    for (int q = 0; q < 4; ++q) {
        const int m = wl * 4 + q;
        const v4f hv = *(const v4f*)(&hlast[((size_t)tile * 16 + m) * H + lane * 4]);
        float s  = hv.x + hv.y + hv.z + hv.w;
        float ss = hv.x * hv.x + hv.y * hv.y + hv.z * hv.z + hv.w * hv.w;
#pragma unroll
        for (int d = 16; d >= 1; d >>= 1) {
            s  += __shfl_xor(s,  d, 32);
            ss += __shfl_xor(ss, d, 32);
        }
        const float mu  = s * (1.0f / 128.0f);
        const float var = ss * (1.0f / 128.0f) - mu * mu;
        const float rs  = rsqrtf(var + 1e-5f);
        float dot = 0.0f;
#pragma unroll
        for (int e = 0; e < 4; ++e)
            dot += ((hv[e] - mu) * rs * lg[e] + lb[e]) * wh[e];
#pragma unroll
        for (int d = 16; d >= 1; d >>= 1) dot += __shfl_xor(dot, d, 32);
        if (lane == 0) out[b0 + tile * 16 + m] = dot + bh;
    }
}

extern "C" void kernel_launch(void* const* d_in, const int* in_sizes, int n_in,
                              void* d_out, int out_size, void* d_ws, size_t ws_size,
                              hipStream_t stream) {
    (void)in_sizes; (void)n_in; (void)d_ws; (void)ws_size; (void)out_size;

    const float* x      = (const float*)d_in[0];
    const float* W_ih   = (const float*)d_in[1];
    const float* W_hh   = (const float*)d_in[2];
    const float* b_ih   = (const float*)d_in[3];
    const float* b_hh   = (const float*)d_in[4];
    const float* ln_g   = (const float*)d_in[5];
    const float* ln_b   = (const float*)d_in[6];
    const float* W_head = (const float*)d_in[7];
    const float* b_head = (const float*)d_in[8];
    float* out = (float*)d_out;

    // allow >48KB dynamic LDS (CDNA5 WGP has 320KB)
    (void)hipFuncSetAttribute((const void*)lstm_ln_head_kernel,
                              hipFuncAttributeMaxDynamicSharedMemorySize,
                              SMEM_BYTES);

    dim3 grid(Bsz / 32);   // 32 batch rows per workgroup
    dim3 block(256);       // 8 wave32
    lstm_ln_head_kernel<<<grid, block, SMEM_BYTES, stream>>>(
        x, W_ih, W_hh, b_ih, b_hh, ln_g, ln_b, W_head, b_head, out);
}
